// SimpleAttention_67224828117506
// MI455X (gfx1250) — matrix operand order repaired and checked by reading, not compiled, not run on hardware
//
#include <hip/hip_runtime.h>
#include <hip/hip_bf16.h>

// ---------------------------------------------------------------------------
// Efficient attention for MI455X (gfx1250, wave32, WMMA bf16 16x16x32).
// Memory-bound (~146 GFLOP vs ~330MB @ 23.3TB/s) -> bf16 intermediates.
// GEMMs: LDS double-buffered, tiles staged with GLOBAL_LOAD_ASYNC_TO_LDS_B128
// using fixed base addresses + 24-bit immediate offsets (no per-step 64-bit
// address VALU), ASYNCcnt + one workgroup barrier per K-step. WMMA fragment
// reads are contiguous ds_load_b128 pairs (A: [M][K], B: [N][K], stride 40).
// ---------------------------------------------------------------------------

typedef __bf16 bf16_t;
typedef __attribute__((ext_vector_type(16))) __bf16 v16bf;
typedef __attribute__((ext_vector_type(8)))  __bf16 v8bf;
typedef __attribute__((ext_vector_type(8)))  float  v8f;

#define DIMC     512
#define HEADS    8
#define DHEAD    64
#define HIDDEN   512
#define NTOK     4096          // h*w = 64*64
#define BATCH    16
#define QKV_ROWS 1536
#define QSCALE   0.125f        // 64^-0.5
#define RMS_EPS  1e-12f
#define SQRT_C   22.62741699796952f   // sqrt(512)
#define LSTR     40            // LDS row stride in bf16 elems (80B, 16B-aligned)

// Async 16B global -> LDS copy; fixed VGPR base + compile-time byte offset.
// Generic pointers to LDS carry the LDS offset in their low 32 bits (ISA 10.2).
static __device__ __forceinline__ void async_cp16(bf16_t* lds_dst, const bf16_t* gbase, int imm) {
  unsigned l = (unsigned)(unsigned long long)lds_dst;
  unsigned long long g = (unsigned long long)gbase;
  asm volatile("global_load_async_to_lds_b128 %0, %1, off offset:%2"
               :: "v"(l), "v"(g), "i"(imm) : "memory");
}
static __device__ __forceinline__ void wait_async() {
#if __has_builtin(__builtin_amdgcn_s_wait_asynccnt)
  __builtin_amdgcn_s_wait_asynccnt(0);
#else
  asm volatile("s_wait_asynccnt 0" ::: "memory");
#endif
}

// ---- WMMA fragment loaders (ISA 7.12.2 layouts, wave32) -------------------
// A fragment from [M][K] tile: lane L -> M = mt*16 + L%16;
//   elems 0..7  = K (L/16)*8 + 0..7 ; elems 8..15 = K (L/16)*8 + 16..23
static __device__ __forceinline__ v16bf load_a_frag(const bf16_t* As, int mt, int lane) {
  const bf16_t* p = As + (mt * 16 + (lane & 15)) * LSTR + ((lane >> 4) << 3);
  v8bf lo = *(const v8bf*)p;
  v8bf hi = *(const v8bf*)(p + 16);
  return __builtin_shufflevector(lo, hi, 0,1,2,3,4,5,6,7,8,9,10,11,12,13,14,15);
}
// B fragment from [N][K] tile: lane L -> N = nt*16 + L%16;
//   elems 0..7  = K (L/16)*8 + 0..7 ; elems 8..15 = K (L/16)*8 + 16..23
static __device__ __forceinline__ v16bf load_b_frag(const bf16_t* Bs, int nt, int lane) {
  const bf16_t* p = Bs + (nt * 16 + (lane & 15)) * LSTR + ((lane >> 4) << 3);
  v8bf lo = *(const v8bf*)p;
  v8bf hi = *(const v8bf*)(p + 16);
  return __builtin_shufflevector(lo, hi, 0,1,2,3,4,5,6,7,8,9,10,11,12,13,14,15);
}

#define WMMA_BF16(a, b, c) \
  __builtin_amdgcn_wmma_f32_16x16x32_bf16(false, (a), false, (b), (short)0, (c), false, false)

// ---- stage 0: weights f32 -> bf16 -----------------------------------------
__global__ __launch_bounds__(256)
void cvt_weights_kernel(const float* __restrict__ Wq, const float* __restrict__ Wo,
                        bf16_t* __restrict__ wq, bf16_t* __restrict__ wo) {
  const int i = blockIdx.x * 256 + threadIdx.x;
  if (i < QKV_ROWS * DIMC) wq[i] = (bf16_t)Wq[i];
  if (i < DIMC * HIDDEN)   wo[i] = (bf16_t)Wo[i];
}

// ---- stage 1: RMSNorm(x) -> xnT[b][n][c] bf16 (contiguous writes) ---------
__global__ __launch_bounds__(256)
void rmsnorm_in_kernel(const float* __restrict__ x, const float* __restrict__ g1,
                       bf16_t* __restrict__ xnT) {
  const int tid = blockIdx.x * 256 + threadIdx.x;      // over BATCH*NTOK
  const int n = tid & (NTOK - 1);
  const int b = tid >> 12;
  const float* xp = x + (size_t)b * DIMC * NTOK + n;
  bf16_t*      op = xnT + ((size_t)b * NTOK + n) * DIMC;
  float ss = 0.f;
  for (int c = 0; c < DIMC; ++c) { float v = xp[(size_t)c * NTOK]; ss += v * v; }
  const float sc = SQRT_C / fmaxf(sqrtf(ss), RMS_EPS);
  for (int c = 0; c < DIMC; ++c)
    op[c] = (bf16_t)(xp[(size_t)c * NTOK] * sc * g1[c]);
}

// ---- stage 2: QKV GEMM: qkv[b][o][n] = Wqkv[o][c] x xnT[b][n][c]^T --------
// 64(M=o) x 128(N=n) tile; wave w: mt=w>>1, 4 B-frags at nt0=(w&1)*4.
__global__ __launch_bounds__(256)
void qkv_gemm_kernel(const bf16_t* __restrict__ W, const bf16_t* __restrict__ XT,
                     bf16_t* __restrict__ Out) {
  __shared__ bf16_t As[2][64 * LSTR];
  __shared__ bf16_t Bs[2][128 * LSTR];
  const int row0 = blockIdx.x * 64, col0 = blockIdx.y * 128, b = blockIdx.z;
  const bf16_t* Xb = XT + (size_t)b * NTOK * DIMC;
  bf16_t*       Ob = Out + (size_t)b * QKV_ROWS * NTOK;
  const int t = threadIdx.x, w = t >> 5, lane = t & 31;
  const int mt = w >> 1, nt0 = (w & 1) * 4;
  const int ar = t >> 2, ako = (t & 3) * 8;       // A build: 64x32
  const int br = t >> 1, bko = (t & 1) * 16;      // B build: 128x32
  const bf16_t* Asrc = W  + (size_t)(row0 + ar) * DIMC + ako;
  const bf16_t* Bsrc = Xb + (size_t)(col0 + br) * DIMC + bko;
  bf16_t* Ad[2] = { &As[0][ar * LSTR + ako], &As[1][ar * LSTR + ako] };
  bf16_t* Bd[2] = { &Bs[0][br * LSTR + bko], &Bs[1][br * LSTR + bko] };
  async_cp16(Ad[0], Asrc, 0);
  async_cp16(Bd[0], Bsrc, 0);
  async_cp16(Bd[0] + 8, Bsrc, 16);
  v8f acc[4] = {{}, {}, {}, {}};
#pragma unroll
  for (int s = 0; s < 16; ++s) {                  // k0 = s*32
    const int buf = s & 1;
    wait_async();
    __syncthreads();
    if (s < 15) {
      async_cp16(Ad[buf ^ 1], Asrc, (s + 1) * 64);
      async_cp16(Bd[buf ^ 1], Bsrc, (s + 1) * 64);
      async_cp16(Bd[buf ^ 1] + 8, Bsrc, (s + 1) * 64 + 16);
    }
    v16bf a = load_a_frag(As[buf], mt, lane);
#pragma unroll
    for (int j = 0; j < 4; ++j) {
      v16bf bf = load_b_frag(Bs[buf], nt0 + j, lane);
      acc[j] = WMMA_BF16(a, bf, acc[j]);
    }
  }
  const int rbase = row0 + mt * 16 + ((lane >> 4) << 3);
#pragma unroll
  for (int j = 0; j < 4; ++j) {
    const int c = col0 + (nt0 + j) * 16 + (lane & 15);
#pragma unroll
    for (int r = 0; r < 8; ++r)
      Ob[(size_t)(rbase + r) * NTOK + c] = (bf16_t)acc[j][r];
  }
}

// ---- stage 3a: softmax(q) over d, *SCALE -> qT[b][h][n][d] ----------------
__global__ __launch_bounds__(256)
void softmax_q_kernel(const bf16_t* __restrict__ QKV, bf16_t* __restrict__ qT) {
  const int tid = blockIdx.x * 256 + threadIdx.x;      // over B*H*N
  const int n = tid & (NTOK - 1);
  const int bh = tid >> 12;
  const int b = bh >> 3, h = bh & 7;
  const bf16_t* q = QKV + ((size_t)b * QKV_ROWS + h * DHEAD) * NTOK + n;
  bf16_t* o = qT + ((size_t)bh * NTOK + n) * DHEAD;
  float vals[64];
  float m = -3.4e38f;
#pragma unroll
  for (int d = 0; d < 64; ++d) { vals[d] = (float)q[(size_t)d * NTOK]; m = fmaxf(m, vals[d]); }
  float s = 0.f;
#pragma unroll
  for (int d = 0; d < 64; ++d) { vals[d] = __expf(vals[d] - m); s += vals[d]; }
  const float inv = QSCALE / s;
#pragma unroll
  for (int d = 0; d < 64; ++d) o[d] = (bf16_t)(vals[d] * inv);
}

// ---- stage 3b: softmax(k) over n=4096, in place ---------------------------
__global__ __launch_bounds__(256)
void softmax_k_kernel(bf16_t* __restrict__ QKV) {
  __shared__ float red[256];
  const int row = blockIdx.x;                // B*H*64 = 8192 rows
  const int b = row >> 9, rem = row & 511;
  bf16_t* k = QKV + ((size_t)b * QKV_ROWS + 512 + rem) * NTOK;
  const int t = threadIdx.x;
  float m = -3.4e38f;
  for (int i = t; i < NTOK; i += 256) m = fmaxf(m, (float)k[i]);
  red[t] = m; __syncthreads();
  for (int s = 128; s > 0; s >>= 1) { if (t < s) red[t] = fmaxf(red[t], red[t + s]); __syncthreads(); }
  m = red[0]; __syncthreads();
  float sum = 0.f;
  for (int i = t; i < NTOK; i += 256) sum += __expf((float)k[i] - m);
  red[t] = sum; __syncthreads();
  for (int s = 128; s > 0; s >>= 1) { if (t < s) red[t] += red[t + s]; __syncthreads(); }
  const float inv = 1.f / red[0];
  for (int i = t; i < NTOK; i += 256) k[i] = (bf16_t)(__expf((float)k[i] - m) * inv);
}

// ---- stage 4: ctxT[e][d] = sum_n v[e][n]*k[d][n]  (A=v, B=k, K=4096) ------
// outer x8 (pointer bump), inner x16 unrolled (immediate offsets).
__global__ __launch_bounds__(256)
void context_kernel(const bf16_t* __restrict__ QKV, bf16_t* __restrict__ CtxT) {
  __shared__ bf16_t As[2][64 * LSTR];
  __shared__ bf16_t Bs[2][64 * LSTR];
  const int bh = blockIdx.x, b = bh >> 3, h = bh & 7;
  const bf16_t* Kp = QKV + ((size_t)b * QKV_ROWS + 512  + h * DHEAD) * NTOK;
  const bf16_t* Vp = QKV + ((size_t)b * QKV_ROWS + 1024 + h * DHEAD) * NTOK;
  const int t = threadIdx.x, w = t >> 5, lane = t & 31;
  const int mt = w >> 1, nt0 = (w & 1) * 2;
  const int r = t >> 2, ko = (t & 3) * 8;         // both tiles: 64x32
  const bf16_t* Ap = Vp + (size_t)r * NTOK + ko;
  const bf16_t* Bp = Kp + (size_t)r * NTOK + ko;
  bf16_t* Ad[2] = { &As[0][r * LSTR + ko], &As[1][r * LSTR + ko] };
  bf16_t* Bd[2] = { &Bs[0][r * LSTR + ko], &Bs[1][r * LSTR + ko] };
  async_cp16(Ad[0], Ap, 0);
  async_cp16(Bd[0], Bp, 0);
  v8f c0 = {}, c1 = {};
  for (int ob = 0; ob < 8; ++ob) {                // 8 x 16 x 32 = 4096 = K
#pragma unroll
    for (int j = 0; j < 16; ++j) {
      const int buf = j & 1;
      wait_async();
      __syncthreads();
      if (j < 15 || ob < 7) {                     // imm 1024 reaches next block
        async_cp16(Ad[buf ^ 1], Ap, (j + 1) * 64);
        async_cp16(Bd[buf ^ 1], Bp, (j + 1) * 64);
      }
      v16bf a  = load_a_frag(As[buf], mt, lane);
      v16bf b0 = load_b_frag(Bs[buf], nt0, lane);
      v16bf b1 = load_b_frag(Bs[buf], nt0 + 1, lane);
      c0 = WMMA_BF16(a, b0, c0);
      c1 = WMMA_BF16(a, b1, c1);
    }
    Ap += 512;   // 16 steps * 32 elems
    Bp += 512;
  }
  bf16_t* C = CtxT + (size_t)bh * 64 * 64;     // [e][d]
  const int rbase = mt * 16 + ((lane >> 4) << 3);
  const int cA = nt0 * 16 + (lane & 15);
#pragma unroll
  for (int rr = 0; rr < 8; ++rr) {
    C[(size_t)(rbase + rr) * 64 + cA]      = (bf16_t)c0[rr];
    C[(size_t)(rbase + rr) * 64 + cA + 16] = (bf16_t)c1[rr];
  }
}

// ---- stage 5: attnT[b][n][h*64+e] = sum_d qT[n][d]*ctxT[e][d] -------------
// 128(M=n) x 64(N=e) tile; wave w: mt=w, 4 B-frags.
__global__ __launch_bounds__(256)
void attn_out_kernel(const bf16_t* __restrict__ qT, const bf16_t* __restrict__ CtxT,
                     bf16_t* __restrict__ AttnT) {
  __shared__ bf16_t As[2][128 * LSTR];
  __shared__ bf16_t Bs[2][64 * LSTR];
  const int n0 = blockIdx.x * 128;
  const int bh = blockIdx.y, b = bh >> 3, h = bh & 7;
  const bf16_t* Qp = qT + (size_t)bh * NTOK * DHEAD;     // [n][d]
  const bf16_t* Cx = CtxT + (size_t)bh * 64 * 64;        // [e][d]
  bf16_t* Ob = AttnT + (size_t)b * NTOK * HIDDEN + h * DHEAD;
  const int t = threadIdx.x, w = t >> 5, lane = t & 31;
  const int mt = w;
  const int ar = t >> 1, ako = (t & 1) * 16;      // A build: 128x32
  const int br = t >> 2, bko = (t & 3) * 8;       // B build: 64x32
  const bf16_t* Asrc = Qp + (size_t)(n0 + ar) * DHEAD + ako;
  const bf16_t* Bsrc = Cx + (size_t)br * 64 + bko;
  bf16_t* Ad[2] = { &As[0][ar * LSTR + ako], &As[1][ar * LSTR + ako] };
  bf16_t* Bd[2] = { &Bs[0][br * LSTR + bko], &Bs[1][br * LSTR + bko] };
  async_cp16(Ad[0], Asrc, 0);
  async_cp16(Ad[0] + 8, Asrc, 16);
  async_cp16(Bd[0], Bsrc, 0);
  v8f acc[4] = {{}, {}, {}, {}};
#pragma unroll
  for (int s = 0; s < 2; ++s) {                   // k0 = s*32
    const int buf = s & 1;
    wait_async();
    __syncthreads();
    if (s < 1) {
      async_cp16(Ad[buf ^ 1], Asrc, 64);
      async_cp16(Ad[buf ^ 1] + 8, Asrc, 80);
      async_cp16(Bd[buf ^ 1], Bsrc, 64);
    }
    v16bf a = load_a_frag(As[buf], mt, lane);
#pragma unroll
    for (int j = 0; j < 4; ++j) {
      v16bf bf = load_b_frag(Bs[buf], j, lane);
      acc[j] = WMMA_BF16(a, bf, acc[j]);
    }
  }
  const int rbase = n0 + mt * 16 + ((lane >> 4) << 3);
#pragma unroll
  for (int j = 0; j < 4; ++j) {
    const int e = j * 16 + (lane & 15);
#pragma unroll
    for (int rr = 0; rr < 8; ++rr)
      Ob[(size_t)(rbase + rr) * HIDDEN + e] = (bf16_t)acc[j][rr];
  }
}

// ---- stage 6: projT[b][n][o] = sum_c attnT[n][c]*Wout[o][c] + bout[o] -----
// 64(M=n) x 128(N=o) tile; f32 coalesced stores.
__global__ __launch_bounds__(256)
void proj_gemm_kernel(const bf16_t* __restrict__ AttnT, const bf16_t* __restrict__ W,
                      const float* __restrict__ bias, float* __restrict__ OutT) {
  __shared__ bf16_t As[2][64 * LSTR];
  __shared__ bf16_t Bs[2][128 * LSTR];
  const int n0 = blockIdx.x * 64, o0 = blockIdx.y * 128, b = blockIdx.z;
  const bf16_t* Ab = AttnT + (size_t)b * NTOK * HIDDEN;
  float*        Ob = OutT + (size_t)b * NTOK * DIMC;
  const int t = threadIdx.x, w = t >> 5, lane = t & 31;
  const int mt = w >> 1, nt0 = (w & 1) * 4;
  const int ar = t >> 2, ako = (t & 3) * 8;       // A build: 64x32
  const int br = t >> 1, bko = (t & 1) * 16;      // B build: 128x32
  const bf16_t* Asrc = Ab + (size_t)(n0 + ar) * HIDDEN + ako;
  const bf16_t* Bsrc = W  + (size_t)(o0 + br) * HIDDEN + bko;
  bf16_t* Ad[2] = { &As[0][ar * LSTR + ako], &As[1][ar * LSTR + ako] };
  bf16_t* Bd[2] = { &Bs[0][br * LSTR + bko], &Bs[1][br * LSTR + bko] };
  async_cp16(Ad[0], Asrc, 0);
  async_cp16(Bd[0], Bsrc, 0);
  async_cp16(Bd[0] + 8, Bsrc, 16);
  v8f acc[4] = {{}, {}, {}, {}};
#pragma unroll
  for (int s = 0; s < 16; ++s) {                  // k0 = s*32
    const int buf = s & 1;
    wait_async();
    __syncthreads();
    if (s < 15) {
      async_cp16(Ad[buf ^ 1], Asrc, (s + 1) * 64);
      async_cp16(Bd[buf ^ 1], Bsrc, (s + 1) * 64);
      async_cp16(Bd[buf ^ 1] + 8, Bsrc, (s + 1) * 64 + 16);
    }
    v16bf a = load_a_frag(As[buf], mt, lane);
#pragma unroll
    for (int j = 0; j < 4; ++j) {
      v16bf bf = load_b_frag(Bs[buf], nt0 + j, lane);
      acc[j] = WMMA_BF16(a, bf, acc[j]);
    }
  }
  const int rbase = n0 + mt * 16 + ((lane >> 4) << 3);
#pragma unroll
  for (int j = 0; j < 4; ++j) {
    const int o = o0 + (nt0 + j) * 16 + (lane & 15);
    const float bb = bias[o];
#pragma unroll
    for (int rr = 0; rr < 8; ++rr)
      Ob[(size_t)(rbase + rr) * DIMC + o] = acc[j][rr] + bb;
  }
}

// ---- stage 7: final RMSNorm (reads projT[b][n][o]) -> d_out[b][c][n] ------
__global__ __launch_bounds__(256)
void rmsnorm_out_kernel(const float* __restrict__ pT, const float* __restrict__ g2,
                        float* __restrict__ out) {
  const int tid = blockIdx.x * 256 + threadIdx.x;      // over BATCH*NTOK
  const int n = tid & (NTOK - 1);
  const int b = tid >> 12;
  const float* xp = pT + ((size_t)b * NTOK + n) * DIMC;   // contiguous
  float*       op = out + (size_t)b * DIMC * NTOK + n;
  float ss = 0.f;
  for (int c = 0; c < DIMC; ++c) { float v = xp[c]; ss += v * v; }
  const float sc = SQRT_C / fmaxf(sqrtf(ss), RMS_EPS);
  for (int c = 0; c < DIMC; ++c)
    op[(size_t)c * NTOK] = xp[c] * sc * g2[c];
}

// ---------------------------------------------------------------------------
extern "C" void kernel_launch(void* const* d_in, const int* in_sizes, int n_in,
                              void* d_out, int out_size, void* d_ws, size_t ws_size,
                              hipStream_t stream) {
  const float* x    = (const float*)d_in[0];
  const float* g1   = (const float*)d_in[1];
  const float* Wqkv = (const float*)d_in[2];
  const float* Wout = (const float*)d_in[3];
  const float* bout = (const float*)d_in[4];
  const float* g2   = (const float*)d_in[5];
  float* out = (float*)d_out;

  char* ws = (char*)d_ws;
  size_t off = 0;
  bf16_t* wqkv_bf = (bf16_t*)(ws + off); off += (size_t)QKV_ROWS * DIMC * sizeof(bf16_t);
  bf16_t* wout_bf = (bf16_t*)(ws + off); off += (size_t)DIMC * HIDDEN * sizeof(bf16_t);
  bf16_t* xnT     = (bf16_t*)(ws + off); off += (size_t)BATCH * NTOK * DIMC * sizeof(bf16_t);
  const size_t qkv_off = off;
  bf16_t* qkv     = (bf16_t*)(ws + off); off += (size_t)BATCH * QKV_ROWS * NTOK * sizeof(bf16_t);
  bf16_t* qT      = (bf16_t*)(ws + off); off += (size_t)BATCH * HEADS * NTOK * DHEAD * sizeof(bf16_t);
  bf16_t* ctxT    = (bf16_t*)(ws + off); off += (size_t)BATCH * HEADS * 64 * 64 * sizeof(bf16_t);
  bf16_t* attnT   = (bf16_t*)(ws + off); off += (size_t)BATCH * NTOK * HIDDEN * sizeof(bf16_t);
  float*  projT   = (float*)(ws + qkv_off);  // reuse qkv region (201MB >= 134MB)

  cvt_weights_kernel<<<(QKV_ROWS * DIMC + 255) / 256, 256, 0, stream>>>(Wqkv, Wout, wqkv_bf, wout_bf);
  rmsnorm_in_kernel<<<(BATCH * NTOK) / 256, 256, 0, stream>>>(x, g1, xnT);
  qkv_gemm_kernel<<<dim3(QKV_ROWS / 64, NTOK / 128, BATCH), 256, 0, stream>>>(wqkv_bf, xnT, qkv);
  softmax_q_kernel<<<(BATCH * HEADS * NTOK) / 256, 256, 0, stream>>>(qkv, qT);
  softmax_k_kernel<<<BATCH * HEADS * DHEAD, 256, 0, stream>>>(qkv);
  context_kernel<<<BATCH * HEADS, 256, 0, stream>>>(qkv, ctxT);
  attn_out_kernel<<<dim3(NTOK / 128, BATCH * HEADS), 256, 0, stream>>>(qT, ctxT, attnT);
  proj_gemm_kernel<<<dim3(NTOK / 64, HIDDEN / 128, BATCH), 256, 0, stream>>>(attnT, wout_bf, bout, projT);
  rmsnorm_out_kernel<<<(BATCH * NTOK) / 256, 256, 0, stream>>>(projT, g2, out);
}
